// StableSplineLinear_4672924418486
// MI455X (gfx1250) — matrix operand order repaired and checked by reading, not compiled, not run on hardware
//
#include <hip/hip_runtime.h>

typedef __attribute__((ext_vector_type(16))) _Float16 v16h;
typedef __attribute__((ext_vector_type(8)))  _Float16 h8;
typedef __attribute__((ext_vector_type(8)))  float    v8f;
typedef __attribute__((ext_vector_type(4)))  float    f4;

union AFrag { v16h v; h8 h[2]; };
union BFrag { v16h v; h8 h[2]; _Float16 e[16]; };

constexpr int IN   = 512;
constexpr int OUT  = 512;
constexpr int T    = 11;
constexpr int BM   = 64;            // rows per block (4 row-tiles)
constexpr int BN   = 128;           // cols per block (8 col-tiles)
constexpr int AST  = 32 + 32 * T;   // 384: halfs per A-panel row per chunk
constexpr int KW   = 16 * AST;      // 6144: total K (16 chunks)

// Cubic B-spline bases on uniform 15-knot grid in [-1,1] (h = 2/14),
// mirroring the reference Cox-de Boor recursion (in-place, ascending j).
__device__ __forceinline__ void bspline11(float xin, float out[11]) {
    const float h = 2.0f / 14.0f;
    float xc = fminf(fmaxf(xin, -2.0f), 2.0f);
    float b[14];
#pragma unroll
    for (int j = 0; j < 14; ++j) {
        float gj  = -1.0f + h * (float)j;
        float gj1 = -1.0f + h * (float)(j + 1);
        b[j] = (xc >= gj && xc < gj1) ? 1.0f : 0.0f;
    }
#pragma unroll
    for (int k = 1; k <= 3; ++k) {
        float inv = 1.0f / (h * (float)k);   // uniform grid: denom = k*h exactly
#pragma unroll
        for (int j = 0; j < 14 - k; ++j) {
            float gj   = -1.0f + h * (float)j;
            float gjk1 = -1.0f + h * (float)(j + k + 1);
            b[j] = (xc - gj) * inv * b[j] + (gjk1 - xc) * inv * b[j + 1];
        }
    }
#pragma unroll
    for (int t = 0; t < 11; ++t) out[t] = b[t];
}

// ---------------------------------------------------------------------------
// Prep: pack base_weight + spline_weight into one f16 matrix W16 (OUT, KW)
// in GEMM consumption order: k = chunk*384 + [ 0..31 : base(i0+e) |
//                                             32..383 : spline flat i0*11+e ].
// Each thread converts 16 contiguous source floats -> 16 contiguous halfs.
// ---------------------------------------------------------------------------
__global__ __launch_bounds__(256) void kan_pack_weights(
    const float* __restrict__ bw, const float* __restrict__ sw,
    _Float16* __restrict__ w16)
{
    const int gid = blockIdx.x * 256 + threadIdx.x;   // OUT * (KW/16) = 512*384
    const int o = gid / (KW / 16);
    const int g = gid - o * (KW / 16);
    const int k = g * 16;
    const int ic = k / AST;
    const int r  = k - ic * AST;
    const float* src = (r < 32)
        ? (bw + (size_t)o * IN   + ic * 32  + r)
        : (sw + (size_t)o * (IN*T) + ic * (32*T) + (r - 32));
    f4 s0 = *(const f4*)(src);
    f4 s1 = *(const f4*)(src + 4);
    f4 s2 = *(const f4*)(src + 8);
    f4 s3 = *(const f4*)(src + 12);
    BFrag b;
#pragma unroll
    for (int e = 0; e < 4; ++e) {
        b.e[e]      = (_Float16)s0[e];
        b.e[4 + e]  = (_Float16)s1[e];
        b.e[8 + e]  = (_Float16)s2[e];
        b.e[12 + e] = (_Float16)s3[e];
    }
    h8* dst = (h8*)(w16 + (size_t)o * KW + k);
    dst[0] = b.h[0];
    dst[1] = b.h[1];
}

// ---------------------------------------------------------------------------
// Main fused GEMM: out[b,o] = sum_k A[b,k] * W16[o,k], K = 6144, where the
// A-panel (x | spline bases) is computed on the fly per 32-feature chunk.
// Wave tiling: 2 row-tiles x 2 col-tiles -> B fragments duplicated only 2x.
// ---------------------------------------------------------------------------
__global__ __launch_bounds__(256) void kan_wmma_packed(
    const float* __restrict__ x, const _Float16* __restrict__ w16,
    float* __restrict__ out, int N)
{
    __shared__ __align__(16) _Float16 As[BM * AST];   // 48 KB

    const int tid  = threadIdx.x;
    const int lane = tid & 31;
    const int w    = tid >> 5;     // wave 0..7
    const int rtp  = w >> 2;       // row-tile pair 0..1 -> row tiles rtp*2+{0,1}
    const int colp = w & 3;        // col-tile pair 0..3 -> col tiles colp*2+{0,1}

    const int b0 = blockIdx.x * BM;
    const int o0 = blockIdx.y * BN;

    const int mlo  = lane & 15;
    const int kb8  = (lane >> 4) * 16;    // A frag: K sub-offset
    const int nB   = lane & 15;
    const int kb16 = (lane >> 4) * 16;    // B frag: K sub-offset

    v8f acc[2][2];
#pragma unroll
    for (int i = 0; i < 2; ++i)
#pragma unroll
        for (int j = 0; j < 2; ++j)
#pragma unroll
            for (int r = 0; r < 8; ++r) acc[i][j][r] = 0.0f;

#pragma unroll 1
    for (int ic = 0; ic < 16; ++ic) {
        const int i0 = ic * 32;
        __syncthreads();

        // ---- stage A panel: 64 rows x (32 x | 32*11 bases) in f16 ----
#pragma unroll 1
        for (int p = 0; p < 8; ++p) {
            const int row = w + p * 8;           // 0..63
            const float xv = x[(size_t)(b0 + row) * IN + (i0 + lane)];
            _Float16* ar = &As[row * AST];
            ar[lane] = (_Float16)xv;             // raw x for base GEMM
            float bs[11];
            bspline11(xv, bs);
            _Float16* br = ar + 32 + lane * T;   // spline K flat = i_local*11 + t
#pragma unroll
            for (int t = 0; t < T; ++t) br[t] = (_Float16)bs[t];
        }
        __syncthreads();

        // ---- 12 WMMA k-steps of 32 (s=0 base, s=1..11 spline) ----
#pragma unroll
        for (int s = 0; s < 12; ++s) {
            const int ko = s * 32;
            AFrag a[2];
#pragma unroll
            for (int i = 0; i < 2; ++i) {
                const int mA = (rtp * 2 + i) * 16 + mlo;
                const _Float16* arow = &As[mA * AST + ko + kb8];
                a[i].h[0] = *(const h8*)(arow);        // K = ko + kb8 + [0..7]
                a[i].h[1] = *(const h8*)(arow + 8);    // K = ko + kb8 + 8 + [0..7]
            }
            BFrag b[2];
#pragma unroll
            for (int j = 0; j < 2; ++j) {
                const int o = o0 + (colp * 2 + j) * 16 + nB;
                const _Float16* wp = w16 + (size_t)o * KW + (ic * AST + ko + kb16);
                b[j].h[0] = *(const h8*)(wp);
                b[j].h[1] = *(const h8*)(wp + 8);
            }
#pragma unroll
            for (int i = 0; i < 2; ++i)
#pragma unroll
                for (int j = 0; j < 2; ++j)
                    acc[i][j] = __builtin_amdgcn_wmma_f32_16x16x32_f16(
                        false, a[i].v, false, b[j].v, (short)0, acc[i][j], false, false);
        }
    }

    // ---- write D (f32 C/D layout: VGPR r -> M = r + 8*(lane>=16), N = lane&15) ----
#pragma unroll
    for (int i = 0; i < 2; ++i) {
        const int mrow = b0 + (rtp * 2 + i) * 16 + 8 * (lane >> 4);
#pragma unroll
        for (int j = 0; j < 2; ++j) {
            const int oc = o0 + (colp * 2 + j) * 16 + nB;
#pragma unroll
            for (int r = 0; r < 8; ++r)
                out[(size_t)(mrow + r) * OUT + oc] = acc[i][j][r];
        }
    }
}

// ---------------------------------------------------------------------------
// Fallback (no workspace): fused f32->f16 weight conversion in the hot loop.
// ---------------------------------------------------------------------------
__global__ __launch_bounds__(256) void kan_wmma_fused(
    const float* __restrict__ x, const float* __restrict__ bw,
    const float* __restrict__ sw, float* __restrict__ out, int N)
{
    __shared__ __align__(16) _Float16 As[BM * AST];

    const int tid  = threadIdx.x;
    const int lane = tid & 31;
    const int w    = tid >> 5;
    const int rtp  = w >> 2;
    const int colp = w & 3;
    const int b0 = blockIdx.x * BM;
    const int o0 = blockIdx.y * BN;
    const int mlo  = lane & 15;
    const int kb8  = (lane >> 4) * 16;
    const int nB   = lane & 15;
    const int kb16 = (lane >> 4) * 16;

    v8f acc[2][2];
#pragma unroll
    for (int i = 0; i < 2; ++i)
#pragma unroll
        for (int j = 0; j < 2; ++j)
#pragma unroll
            for (int r = 0; r < 8; ++r) acc[i][j][r] = 0.0f;

#pragma unroll 1
    for (int ic = 0; ic < 16; ++ic) {
        const int i0 = ic * 32;
        __syncthreads();
#pragma unroll 1
        for (int p = 0; p < 8; ++p) {
            const int row = w + p * 8;
            const float xv = x[(size_t)(b0 + row) * IN + (i0 + lane)];
            _Float16* ar = &As[row * AST];
            ar[lane] = (_Float16)xv;
            float bs[11];
            bspline11(xv, bs);
            _Float16* br = ar + 32 + lane * T;
#pragma unroll
            for (int t = 0; t < T; ++t) br[t] = (_Float16)bs[t];
        }
        __syncthreads();

#pragma unroll
        for (int s = 0; s < 12; ++s) {
            const int ko = s * 32;
            AFrag a[2];
#pragma unroll
            for (int i = 0; i < 2; ++i) {
                const int mA = (rtp * 2 + i) * 16 + mlo;
                const _Float16* arow = &As[mA * AST + ko + kb8];
                a[i].h[0] = *(const h8*)(arow);
                a[i].h[1] = *(const h8*)(arow + 8);
            }
            BFrag b[2];
#pragma unroll
            for (int j = 0; j < 2; ++j) {
                const int o = o0 + (colp * 2 + j) * 16 + nB;
                const float* wp = (s == 0)
                    ? (bw + (size_t)o * IN + (i0 + kb16))
                    : (sw + (size_t)o * (IN*T) + ((size_t)i0 * T + (s - 1) * 32 + kb16));
                f4 w0 = *(const f4*)(wp);
                f4 w1 = *(const f4*)(wp + 4);
                f4 w2 = *(const f4*)(wp + 8);
                f4 w3 = *(const f4*)(wp + 12);
#pragma unroll
                for (int e = 0; e < 4; ++e) {
                    b[j].e[e]      = (_Float16)w0[e];
                    b[j].e[4 + e]  = (_Float16)w1[e];
                    b[j].e[8 + e]  = (_Float16)w2[e];
                    b[j].e[12 + e] = (_Float16)w3[e];
                }
            }
#pragma unroll
            for (int i = 0; i < 2; ++i)
#pragma unroll
                for (int j = 0; j < 2; ++j)
                    acc[i][j] = __builtin_amdgcn_wmma_f32_16x16x32_f16(
                        false, a[i].v, false, b[j].v, (short)0, acc[i][j], false, false);
        }
    }

#pragma unroll
    for (int i = 0; i < 2; ++i) {
        const int mrow = b0 + (rtp * 2 + i) * 16 + 8 * (lane >> 4);
#pragma unroll
        for (int j = 0; j < 2; ++j) {
            const int oc = o0 + (colp * 2 + j) * 16 + nB;
#pragma unroll
            for (int r = 0; r < 8; ++r)
                out[(size_t)(mrow + r) * OUT + oc] = acc[i][j][r];
        }
    }
}

extern "C" void kernel_launch(void* const* d_in, const int* in_sizes, int n_in,
                              void* d_out, int out_size, void* d_ws, size_t ws_size,
                              hipStream_t stream) {
    const float* x  = (const float*)d_in[0];  // (8192, 512)
    const float* bw = (const float*)d_in[1];  // (512, 512)
    const float* sw = (const float*)d_in[2];  // (512, 512, 11)
    float* out = (float*)d_out;               // (8192, 512)
    const int N = in_sizes[0] / IN;
    dim3 grid(N / BM, OUT / BN);

    const size_t need = (size_t)OUT * KW * sizeof(_Float16);   // 6.29 MB
    if (ws_size >= need) {
        _Float16* w16 = (_Float16*)d_ws;
        kan_pack_weights<<<dim3((OUT * (KW / 16)) / 256), dim3(256), 0, stream>>>(bw, sw, w16);
        kan_wmma_packed<<<grid, dim3(256), 0, stream>>>(x, w16, out, N);
    } else {
        kan_wmma_fused<<<grid, dim3(256), 0, stream>>>(x, bw, sw, out, N);
    }
}
